// TimeIndependentFFSA_38835094291104
// MI455X (gfx1250) — compile-verified, matrix operand order checked by reading, not run on hardware
//
#include <hip/hip_runtime.h>

typedef __attribute__((ext_vector_type(16))) _Float16 v16h;
typedef __attribute__((ext_vector_type(8)))  float    v8f;

#define BB 8192
#define DD 128
#define SS 512
#define TT 2048

// ---------------------------------------------------------------------------
// Pre-pass 1: pack input [B,D] f32 -> f16 tiles in WMMA A-operand lane order.
// Tile = 16 rows (M) x 32 cols (K). tile id = mt*(D/32)+kt. Per tile: 32 lanes
// x 16 halfs, contiguous (32B per lane -> b128 pair load in main kernel).
// Lane L (<16): M=L,  K in {0..7, 16..23}; lane L (>=16): M=L-16, K += 8.
// Element e -> K = (e&7) + (e&8 ? 16 : 0) + (L>=16 ? 8 : 0).
// ---------------------------------------------------------------------------
__global__ void pack_a_kernel(const float* __restrict__ inp,
                              _Float16* __restrict__ a_pk) {
    int t    = blockIdx.x * blockDim.x + threadIdx.x;   // one thread per (tile,lane)
    int lane = t & 31;
    int tile = t >> 5;                                  // mt*4 + kt
    int kt   = tile & 3;
    int mt   = tile >> 2;
    if (mt >= BB / 16) return;
    int m   = lane & 15;
    int khl = (lane >> 4) << 3;                         // 0 or 8
    const float* src = inp + (size_t)(mt * 16 + m) * DD + kt * 32 + khl;
    v16h o;
#pragma unroll
    for (int e = 0; e < 16; ++e) {
        int k = (e & 7) + ((e & 8) ? 16 : 0);
        o[e] = (_Float16)src[k];
    }
    *(v16h*)(a_pk + (size_t)t * 16) = o;
}

// ---------------------------------------------------------------------------
// Pre-pass 2: pack W [D,T] f32 -> f16 tiles in WMMA B-operand lane order.
// B tile = 32 (K) x 16 (N), column-striped: lane L holds column N=L&15 with
// the same K subset rule as A. tile id = nt*(D/32)+kt (so kt is innermost).
// ---------------------------------------------------------------------------
__global__ void pack_w_kernel(const float* __restrict__ W,
                              _Float16* __restrict__ w_pk) {
    int t    = blockIdx.x * blockDim.x + threadIdx.x;
    int lane = t & 31;
    int tile = t >> 5;                                  // nt*4 + kt
    int kt   = tile & 3;
    int nt   = tile >> 2;
    if (nt >= TT / 16) return;
    int n   = nt * 16 + (lane & 15);
    int khl = (lane >> 4) << 3;
    v16h o;
#pragma unroll
    for (int e = 0; e < 16; ++e) {
        int k = kt * 32 + khl + (e & 7) + ((e & 8) ? 16 : 0);
        o[e] = (_Float16)W[(size_t)k * TT + n];
    }
    *(v16h*)(w_pk + (size_t)t * 16) = o;
}

// ---------------------------------------------------------------------------
// Main fused kernel: one block (8 waves) per 16-row batch tile.
//  - WMMA GEMM (16x16x32 f16, f32 acc) for logits z = x@W
//  - per element: log(1 - sigmoid(z + b)) = -softplus(z + b)
//    = fma(-ln2, v_log_f32(1 + exp(-|z|)), -max(z,0))  [raw v_exp/v_log]
//  - ds_add_f32 LDS atomics segment-sum into [16 x 512] from/to accumulators
//  - epilogue: out = (1 - (1-act)*exp(sum_to)) * exp(sum_from)
// ---------------------------------------------------------------------------
__global__ __launch_bounds__(256)
void ffsa_main_kernel(const _Float16* __restrict__ a_pk,
                      const _Float16* __restrict__ w_pk,
                      const float* __restrict__ bias,
                      const int*   __restrict__ from_idx,
                      const int*   __restrict__ to_idx,
                      const float* __restrict__ acts,
                      float* __restrict__ out) {
    __shared__ float s_from[16 * SS];   // 32 KB
    __shared__ float s_to[16 * SS];     // 32 KB

    const int tid  = threadIdx.x;
    const int lane = tid & 31;
    const int wave = tid >> 5;
    const int mt   = blockIdx.x;        // 16-row batch tile

    for (int i = tid; i < 16 * SS; i += 256) {
        s_from[i] = 0.0f;
        s_to[i]   = 0.0f;
    }
    __syncthreads();

    // A operands for this row tile (shared by all N-tiles of this wave).
    v16h a_op[4];
#pragma unroll
    for (int kt = 0; kt < 4; ++kt)
        a_op[kt] = *(const v16h*)(a_pk + ((size_t)(mt * 4 + kt) * 32 + lane) * 16);

    const int colsub  = lane & 15;        // N within tile for this lane's acc
    const int rowbase = (lane >> 4) * 8;  // acc VGPR v holds row rowbase+v

    // Each wave covers 16 of the 128 N-tiles.
#pragma unroll 1
    for (int j = 0; j < 16; ++j) {
        const int nt = wave * 16 + j;
        v8f acc = {};
#pragma unroll
        for (int kt = 0; kt < 4; ++kt) {
            v16h b_op = *(const v16h*)(w_pk + ((size_t)(nt * 4 + kt) * 32 + lane) * 16);
            acc = __builtin_amdgcn_wmma_f32_16x16x32_f16(
                /*neg_a=*/false, a_op[kt], /*neg_b=*/false, b_op,
                /*c_mod=*/(short)0, acc, /*reuse_a=*/false, /*reuse_b=*/false);
        }

        const int   n  = nt * 16 + colsub;
        const float bn = bias[n];
        const int   fs = from_idx[n];
        const int   ts = to_idx[n];

#pragma unroll
        for (int v = 0; v < 8; ++v) {
            const float z = acc[v] + bn;
            // log(1 - sigmoid(z)) = -softplus(z) = -max(z,0) - log(1+exp(-|z|)).
            // Log argument is in (1,2]: call raw v_log_f32 (log2) directly and
            // fold the ln2 scale + max-term into one v_fma_f32. Avoids the
            // libm denormal-guard cmp/cndmask/ldexp sequence.
            const float ez = __expf(-fabsf(z));                    // v_exp_f32
            const float l2 = __builtin_amdgcn_logf(1.0f + ez);     // v_log_f32
            const float val = fmaf(-0.69314718055994531f, l2, -fmaxf(z, 0.0f));
            const int row = rowbase + v;
            atomicAdd(&s_from[row * SS + fs], val);   // ds_add_f32
            atomicAdd(&s_to[row * SS + ts], val);     // ds_add_f32
        }
    }
    __syncthreads();

    // Epilogue: fuzzy OR/AND combine, fully fused.
    const float* actp = acts + (size_t)mt * 16 * SS;
    float*       outp = out + (size_t)mt * 16 * SS;
    for (int i = tid; i < 16 * SS; i += 256) {
        const float a    = actp[i];
        const float e_to = __expf(s_to[i]);     // = 1 - to_act
        const float e_fr = __expf(s_from[i]);   // = 1 - from_act
        outp[i] = (1.0f - (1.0f - a) * e_to) * e_fr;
    }
}

extern "C" void kernel_launch(void* const* d_in, const int* in_sizes, int n_in,
                              void* d_out, int out_size, void* d_ws, size_t ws_size,
                              hipStream_t stream) {
    const float* input    = (const float*)d_in[0];   // [B, D]
    const float* acts     = (const float*)d_in[1];   // [B, S]
    const float* W        = (const float*)d_in[2];   // [D, T]
    const float* bias     = (const float*)d_in[3];   // [T]
    const int*   from_idx = (const int*)d_in[4];     // [T]
    const int*   to_idx   = (const int*)d_in[5];     // [T]
    float*       out      = (float*)d_out;           // [B, S]

    // Workspace: packed f16 A (2 MB) followed by packed f16 W (512 KB).
    _Float16* a_pk = (_Float16*)d_ws;
    _Float16* w_pk = a_pk + (size_t)BB * DD;

    // Pack A: (B/16)*(D/32) tiles * 32 lanes = 65536 threads.
    pack_a_kernel<<<(BB / 16) * (DD / 32) * 32 / 256, 256, 0, stream>>>(input, a_pk);
    // Pack W: (T/16)*(D/32) tiles * 32 lanes = 16384 threads.
    pack_w_kernel<<<(TT / 16) * (DD / 32) * 32 / 256, 256, 0, stream>>>(W, w_pk);

    // Fused GEMM + segment-sum + fuzzy combine: one block per 16 batch rows.
    ffsa_main_kernel<<<BB / 16, 256, 0, stream>>>(a_pk, w_pk, bias, from_idx,
                                                  to_idx, acts, out);
}
